// MoEFFN_55284819034464
// MI455X (gfx1250) — matrix operand order repaired and checked by reading, not compiled, not run on hardware
//
#include <hip/hip_runtime.h>
#include <hip/hip_bf16.h>
#include <math.h>

#define D_MODEL   1024
#define D_EXPERT  1024
#define NUM_EXP   8
#define N_TOK     8192      // 4 * 2048
#define BM        64
#define BN        128
#define BK        64        // two WMMA k-steps per LDS stage
#define KP        72        // BK + 8 halfs pad; row stride 144B (16B aligned)

typedef __attribute__((ext_vector_type(16))) _Float16 v16h;
typedef __attribute__((ext_vector_type(8)))  _Float16 v8h;
typedef __attribute__((ext_vector_type(4)))  _Float16 v4h;
typedef __attribute__((ext_vector_type(8)))  float    v8f;
typedef unsigned int u32;
typedef __attribute__((ext_vector_type(4))) u32 u32x4;
typedef __attribute__((ext_vector_type(8))) u32 u32x8;

__device__ __forceinline__ int imin(int a, int b) { return a < b ? a : b; }

// ---------------------------------------------------------------------------
// CDNA5 async global->LDS 16B copy (ASYNCcnt; ISA §15.18.3 op 98)
// Generic __shared__ pointer low 32 bits == LDS byte offset (aperture rule).
// ---------------------------------------------------------------------------
__device__ __forceinline__ void cp16_async(void* lds_dst, const void* gsrc) {
  unsigned loff = (unsigned)(size_t)lds_dst;
  asm volatile("global_load_async_to_lds_b128 %0, %1, off"
               :: "v"(loff), "v"(gsrc) : "memory");
}

// ---------------------------------------------------------------------------
// CDNA5 Tensor Data Mover: 2D f16 tile (128 rows x 64 halfs, tensor row
// stride 1024 halfs) -> LDS with HW padding (+4 DWORDs after each 32 DWORDs,
// reproducing the KP=72 pitch).  D# per ISA §8.3/8.4; TENSORcnt-tracked.
// ---------------------------------------------------------------------------
__device__ __forceinline__ void tdm_load_tile(unsigned lds_addr,
                                              const _Float16* gtile) {
  unsigned long long ga = (unsigned long long)(size_t)gtile;
  u32x4 g0;
  g0[0] = 1u;                                           // count=1, user mode
  g0[1] = lds_addr;                                     // lds_addr
  g0[2] = (u32)ga;                                      // global_addr[31:0]
  g0[3] = (u32)((ga >> 32) & 0x1FFFFFFu) | (2u << 30);  // addr[56:32], type=2
  u32x8 g1;
  g1[0] = (1u << 16)      // data_size = 2B
        | (1u << 20)      // pad_enable
        | (4u << 22)      // pad_interval: 32 DWORDs (=128B row)
        | (3u << 25);     // pad_amount: 4 DWORDs (=16B pad)
  g1[1] = (1024u & 0xFFFFu) << 16;     // tensor_dim0 lo16 (K extent)
  g1[2] = (1024u & 0xFFFFu) << 16;     // dim0 hi=0 | tensor_dim1 lo16 (rows)
  g1[3] = (64u << 16);                 // dim1 hi=0 | tile_dim0 = 64 halfs
  g1[4] = 128u;                        // tile_dim1 = 128 rows, tile_dim2 = 0
  g1[5] = 1024u;                       // tensor_dim0_stride lo32 (halfs)
  g1[6] = 0u;                          // stride hi | dim1_stride lo (unused 2D)
  g1[7] = 0u;
  u32x4 gz; gz[0] = 0; gz[1] = 0; gz[2] = 0; gz[3] = 0;
  asm volatile("tensor_load_to_lds %0, %1, %2, %3"
               :: "s"(g0), "s"(g1), "s"(gz), "s"(gz) : "memory");
}

// ---------------------------------------------------------------------------
// 1) fp32 -> f16 activation conversion
// ---------------------------------------------------------------------------
__global__ __launch_bounds__(256) void convert_x_kernel(
    const float* __restrict__ x, _Float16* __restrict__ xh) {
  size_t i = ((size_t)blockIdx.x * 256 + threadIdx.x) * 4;
  float4 v = *(const float4*)(x + i);
  v4h h;
  h[0] = (_Float16)v.x; h[1] = (_Float16)v.y;
  h[2] = (_Float16)v.z; h[3] = (_Float16)v.w;
  *(v4h*)(xh + i) = h;
}

// ---------------------------------------------------------------------------
// 2) Weight convert + transpose: W fp32 [E][K][N] -> WT f16 [E][N][K]
// ---------------------------------------------------------------------------
__global__ __launch_bounds__(256) void convert_wT_kernel(
    const float* __restrict__ W, _Float16* __restrict__ WT) {
  const int e = blockIdx.z;
  const float* We = W  + (size_t)e * D_MODEL * D_EXPERT;
  _Float16*  WTe  = WT + (size_t)e * D_MODEL * D_EXPERT;
  const int kb = blockIdx.x * 32;
  const int nb = blockIdx.y * 32;
  __shared__ float tile[32][33];
  const int tid = threadIdx.x;
  {
    const int r = tid >> 3, c = (tid & 7) * 4;
    float4 v = *(const float4*)(We + (size_t)(kb + r) * D_EXPERT + nb + c);
    tile[r][c + 0] = v.x; tile[r][c + 1] = v.y;
    tile[r][c + 2] = v.z; tile[r][c + 3] = v.w;
  }
  __syncthreads();
  {
    const int n = tid >> 3, k = (tid & 7) * 4;
    v4h h;
    h[0] = (_Float16)tile[k + 0][n];
    h[1] = (_Float16)tile[k + 1][n];
    h[2] = (_Float16)tile[k + 2][n];
    h[3] = (_Float16)tile[k + 3][n];
    *(v4h*)(WTe + (size_t)(nb + n) * D_MODEL + kb + k) = h;
  }
}

// ---------------------------------------------------------------------------
// 3) Router: wave-per-token logits, top-2, renormalized softmax weights
// ---------------------------------------------------------------------------
__global__ __launch_bounds__(256) void router_kernel(
    const float* __restrict__ x, const float* __restrict__ Wr,
    int* __restrict__ counts, int* __restrict__ list, float* __restrict__ wts) {
  const int token = blockIdx.x * 8 + (threadIdx.x >> 5);
  const int lane  = threadIdx.x & 31;
  const float* xr = x + (size_t)token * D_MODEL;
  float acc[NUM_EXP];
#pragma unroll
  for (int e = 0; e < NUM_EXP; ++e) acc[e] = 0.f;
  for (int c = lane; c < D_MODEL; c += 32) {
    float xv = xr[c];
    const float* wr = Wr + c * NUM_EXP;
#pragma unroll
    for (int e = 0; e < NUM_EXP; ++e) acc[e] += xv * wr[e];
  }
#pragma unroll
  for (int e = 0; e < NUM_EXP; ++e) {
#pragma unroll
    for (int m = 16; m >= 1; m >>= 1) acc[e] += __shfl_xor(acc[e], m, 32);
  }
  if (lane == 0) {
    int e0 = 0; float l0 = acc[0];
#pragma unroll
    for (int e = 1; e < NUM_EXP; ++e) if (acc[e] > l0) { l0 = acc[e]; e0 = e; }
    int e1 = -1; float l1 = -3.4e38f;
#pragma unroll
    for (int e = 0; e < NUM_EXP; ++e)
      if (e != e0 && acc[e] > l1) { l1 = acc[e]; e1 = e; }
    float p0 = 1.f / (1.f + __expf(l1 - l0));   // renormalized top-2 softmax
    float p1 = 1.f - p0;
    int s0 = atomicAdd(&counts[e0], 1);
    list[e0 * N_TOK + s0] = token * 2 + 0;
    wts [e0 * N_TOK + s0] = p0;
    int s1 = atomicAdd(&counts[e1], 1);
    list[e1 * N_TOK + s1] = token * 2 + 1;
    wts [e1 * N_TOK + s1] = p1;
  }
}

// ---------------------------------------------------------------------------
// WMMA fragment loads from LDS (layouts per ISA §7.12.2)
// ---------------------------------------------------------------------------
__device__ __forceinline__ v16h load_a_frag(const _Float16* As, int mrow,
                                            int lane, int ks) {
  const int r  = mrow + (lane & 15);
  const int cb = ks + (lane >> 4) * 8;
  union { v16h v; v8h h[2]; } u;
  u.h[0] = *(const v8h*)(As + r * KP + cb);
  u.h[1] = *(const v8h*)(As + r * KP + cb + 16);
  return u.v;
}

__device__ __forceinline__ v16h load_b_frag(const _Float16* Bs, int ncol,
                                            int lane, int ks) {
  const int c  = ncol + (lane & 15);
  const int kb = ks + (lane >> 4) * 8;
  union { v16h v; v8h h[2]; } u;
  u.h[0] = *(const v8h*)(Bs + c * KP + kb);
  u.h[1] = *(const v8h*)(Bs + c * KP + kb + 16);
  return u.v;
}

// ---------------------------------------------------------------------------
// 4) GEMM phase 1: H = silu(Xe @ W1) * (Xe @ W3)
//    Double-buffered pipeline: A via per-lane async B128, B via TDM tiles.
// ---------------------------------------------------------------------------
__global__ __launch_bounds__(256) void gemm13_kernel(
    const _Float16* __restrict__ xh, const _Float16* __restrict__ w1T,
    const _Float16* __restrict__ w3T, const int* __restrict__ counts,
    const int* __restrict__ list, _Float16* __restrict__ H) {
  const int e   = blockIdx.z;
  const int cnt = counts[e];
  const int m0  = blockIdx.x * BM;
  if (m0 >= cnt) return;
  const int n0    = blockIdx.y * BN;
  const int valid = imin(BM, cnt - m0);

  __shared__ __attribute__((aligned(16))) _Float16 As [2][BM * KP];
  __shared__ __attribute__((aligned(16))) _Float16 B1s[2][BN * KP];
  __shared__ __attribute__((aligned(16))) _Float16 B3s[2][BN * KP];
  __shared__ int toks[BM];

  const int tid  = threadIdx.x;
  const int lane = tid & 31, wave = tid >> 5;
  if (tid < BM) toks[tid] = list[e * N_TOK + m0 + imin(tid, valid - 1)];
  __syncthreads();

  const _Float16* w1e = w1T + (size_t)e * D_EXPERT * D_MODEL;
  const _Float16* w3e = w3T + (size_t)e * D_EXPERT * D_MODEL;

  v8f acc1[4], acc3[4];
#pragma unroll
  for (int t = 0; t < 4; ++t)
#pragma unroll
    for (int j = 0; j < 8; ++j) { acc1[t][j] = 0.f; acc3[t][j] = 0.f; }

  const int mrow  = (wave & 3) * 16;
  const int nbase = (wave >> 2) * 64;
  const int arow  = tid >> 3, aoff = (tid & 7) * 8;   // 512 A-chunks, 2/thread

  auto issue_stage = [&](int k0, int buf) {
#pragma unroll
    for (int r = 0; r < 2; ++r) {
      const int c = tid + 256 * r, row = c >> 3, off = (c & 7) * 8;
      cp16_async(&As[buf][row * KP + off],
                 xh + (size_t)(toks[row] >> 1) * D_MODEL + k0 + off);
    }
    if (wave == 0) {
      tdm_load_tile((unsigned)(size_t)&B1s[buf][0],
                    w1e + (size_t)n0 * D_MODEL + k0);
      tdm_load_tile((unsigned)(size_t)&B3s[buf][0],
                    w3e + (size_t)n0 * D_MODEL + k0);
    }
  };

  const int nstages = D_MODEL / BK;   // 16
  issue_stage(0, 0);
  for (int s = 0; s < nstages; ++s) {
    const int cur = s & 1;
    if (s + 1 < nstages) {
      issue_stage((s + 1) * BK, cur ^ 1);
      asm volatile("s_wait_asynccnt 0x2" ::: "memory");
      __builtin_amdgcn_s_wait_tensorcnt(2);
    } else {
      asm volatile("s_wait_asynccnt 0x0" ::: "memory");
      __builtin_amdgcn_s_wait_tensorcnt(0);
    }
    __syncthreads();
#pragma unroll
    for (int ks = 0; ks < BK; ks += 32) {
      v16h a = load_a_frag(As[cur], mrow, lane, ks);
      v16h b1f[4], b3f[4];
#pragma unroll
      for (int t = 0; t < 4; ++t) {
        b1f[t] = load_b_frag(B1s[cur], nbase + t * 16, lane, ks);
        b3f[t] = load_b_frag(B3s[cur], nbase + t * 16, lane, ks);
      }
#pragma unroll
      for (int t = 0; t < 4; ++t) {
        acc1[t] = __builtin_amdgcn_wmma_f32_16x16x32_f16(
            false, a, false, b1f[t], (short)0, acc1[t], false, false);
        acc3[t] = __builtin_amdgcn_wmma_f32_16x16x32_f16(
            false, a, false, b3f[t], (short)0, acc3[t], false, false);
      }
    }
    __syncthreads();
  }

#pragma unroll
  for (int t = 0; t < 4; ++t) {
    const int lcol = nbase + t * 16 + (lane & 15);
#pragma unroll
    for (int j = 0; j < 8; ++j) {
      const int lrow = mrow + j + ((lane >> 4) << 3);   // C/D: VGPR j -> M=j/j+8
      if (lrow < valid) {
        float g = acc1[t][j];
        float hval = g * __builtin_amdgcn_rcpf(1.f + __expf(-g)) * acc3[t][j];
        H[(size_t)toks[lrow] * D_EXPERT + n0 + lcol] = (_Float16)hval;
      }
    }
  }
}

// ---------------------------------------------------------------------------
// 5) GEMM phase 2: Y = H @ W2, scaled by routing weight.
//    rank0 -> d_out, rank1 -> Y1 scratch (race-free, deterministic)
// ---------------------------------------------------------------------------
__global__ __launch_bounds__(256) void gemm2_kernel(
    const _Float16* __restrict__ H, const _Float16* __restrict__ w2T,
    const int* __restrict__ counts, const int* __restrict__ list,
    const float* __restrict__ wts, float* __restrict__ out,
    float* __restrict__ Y1) {
  const int e   = blockIdx.z;
  const int cnt = counts[e];
  const int m0  = blockIdx.x * BM;
  if (m0 >= cnt) return;
  const int n0    = blockIdx.y * BN;
  const int valid = imin(BM, cnt - m0);

  __shared__ __attribute__((aligned(16))) _Float16 As[2][BM * KP];
  __shared__ __attribute__((aligned(16))) _Float16 Bs[2][BN * KP];
  __shared__ int   toks[BM];
  __shared__ float wv[BM];

  const int tid  = threadIdx.x;
  const int lane = tid & 31, wave = tid >> 5;
  if (tid < BM) {
    int s = imin(tid, valid - 1);
    toks[tid] = list[e * N_TOK + m0 + s];
    wv[tid]   = wts [e * N_TOK + m0 + s];
  }
  __syncthreads();

  const _Float16* w2e = w2T + (size_t)e * D_MODEL * D_EXPERT;

  v8f acc[4];
#pragma unroll
  for (int t = 0; t < 4; ++t)
#pragma unroll
    for (int j = 0; j < 8; ++j) acc[t][j] = 0.f;

  const int mrow  = (wave & 3) * 16;
  const int nbase = (wave >> 2) * 64;

  auto issue_stage = [&](int k0, int buf) {
#pragma unroll
    for (int r = 0; r < 2; ++r) {
      const int c = tid + 256 * r, row = c >> 3, off = (c & 7) * 8;
      cp16_async(&As[buf][row * KP + off],
                 H + (size_t)toks[row] * D_EXPERT + k0 + off);
    }
    if (wave == 0) {
      tdm_load_tile((unsigned)(size_t)&Bs[buf][0],
                    w2e + (size_t)n0 * D_EXPERT + k0);
    }
  };

  const int nstages = D_EXPERT / BK;   // 16
  issue_stage(0, 0);
  for (int s = 0; s < nstages; ++s) {
    const int cur = s & 1;
    if (s + 1 < nstages) {
      issue_stage((s + 1) * BK, cur ^ 1);
      asm volatile("s_wait_asynccnt 0x2" ::: "memory");
      __builtin_amdgcn_s_wait_tensorcnt(1);
    } else {
      asm volatile("s_wait_asynccnt 0x0" ::: "memory");
      __builtin_amdgcn_s_wait_tensorcnt(0);
    }
    __syncthreads();
#pragma unroll
    for (int ks = 0; ks < BK; ks += 32) {
      v16h a = load_a_frag(As[cur], mrow, lane, ks);
      v16h bf[4];
#pragma unroll
      for (int t = 0; t < 4; ++t)
        bf[t] = load_b_frag(Bs[cur], nbase + t * 16, lane, ks);
#pragma unroll
      for (int t = 0; t < 4; ++t)
        acc[t] = __builtin_amdgcn_wmma_f32_16x16x32_f16(
            false, a, false, bf[t], (short)0, acc[t], false, false);
    }
    __syncthreads();
  }

#pragma unroll
  for (int t = 0; t < 4; ++t) {
    const int lcol = nbase + t * 16 + (lane & 15);
#pragma unroll
    for (int j = 0; j < 8; ++j) {
      const int lrow = mrow + j + ((lane >> 4) << 3);
      if (lrow < valid) {
        const int t2    = toks[lrow];
        const int token = t2 >> 1;
        float val = acc[t][j] * wv[lrow];
        size_t addr = (size_t)token * D_MODEL + n0 + lcol;
        if ((t2 & 1) == 0) out[addr] = val;
        else               Y1[addr]  = val;
      }
    }
  }
}

// ---------------------------------------------------------------------------
// 6) Combine: out += Y1 (fixed order -> deterministic)
// ---------------------------------------------------------------------------
__global__ __launch_bounds__(256) void combine_kernel(
    float* __restrict__ out, const float* __restrict__ Y1) {
  size_t i = ((size_t)blockIdx.x * 256 + threadIdx.x) * 4;
  float4 o = *(const float4*)(out + i);
  float4 y = *(const float4*)(Y1 + i);
  o.x += y.x; o.y += y.y; o.z += y.z; o.w += y.w;
  *(float4*)(out + i) = o;
}

// ---------------------------------------------------------------------------
extern "C" void kernel_launch(void* const* d_in, const int* in_sizes, int n_in,
                              void* d_out, int out_size, void* d_ws, size_t ws_size,
                              hipStream_t stream) {
  const float* x  = (const float*)d_in[0];
  const float* Wr = (const float*)d_in[1];
  const float* W1 = (const float*)d_in[2];
  const float* W2 = (const float*)d_in[3];
  const float* W3 = (const float*)d_in[4];
  float* out = (float*)d_out;

  // workspace layout
  char* ws = (char*)d_ws;
  const size_t WMAT = (size_t)NUM_EXP * D_MODEL * D_EXPERT;   // 8M elems
  int*   counts = (int*)ws;                                   // 256 B
  int*   list   = (int*)(ws + 256);                           // 256 KB
  float* wts    = (float*)(ws + 256 + (NUM_EXP * N_TOK * 4)); // 256 KB
  _Float16* xh  = (_Float16*)(ws + 256 + 2 * (NUM_EXP * N_TOK * 4));  // 16 MB
  _Float16* w1T = xh  + (size_t)N_TOK * D_MODEL;              // 16 MB
  _Float16* w3T = w1T + WMAT;                                 // 16 MB
  _Float16* w2T = w3T + WMAT;                                 // 16 MB
  _Float16* Hb  = w2T + WMAT;                                 // 32 MB
  float*    Y1  = (float*)(Hb + (size_t)2 * N_TOK * D_EXPERT);// 32 MB

  hipMemsetAsync(counts, 0, 8 * sizeof(int), stream);

  const size_t NC = (size_t)N_TOK * D_MODEL;                  // 8,388,608
  convert_x_kernel<<<NC / (256 * 4), 256, 0, stream>>>(x, xh);
  router_kernel<<<N_TOK / 8, 256, 0, stream>>>(x, Wr, counts, list, wts);

  dim3 gc(D_MODEL / 32, D_EXPERT / 32, NUM_EXP);              // (32, 32, 8)
  convert_wT_kernel<<<gc, 256, 0, stream>>>(W1, w1T);
  convert_wT_kernel<<<gc, 256, 0, stream>>>(W3, w3T);
  convert_wT_kernel<<<gc, 256, 0, stream>>>(W2, w2T);

  dim3 g(N_TOK / BM, D_EXPERT / BN, NUM_EXP);                 // (128, 8, 8)
  gemm13_kernel<<<g, 256, 0, stream>>>(xh, w1T, w3T, counts, list, Hb);
  gemm2_kernel <<<g, 256, 0, stream>>>(Hb, w2T, counts, list, wts, out, Y1);

  combine_kernel<<<NC / (256 * 4), 256, 0, stream>>>(out, Y1);
}
